// PSIBlock_SqrtNorm_47931835024035
// MI455X (gfx1250) — compile-verified, matrix operand order checked by reading, not run on hardware
//
#include <hip/hip_runtime.h>
#include <hip/hip_bf16.h>
#include <math.h>

typedef __attribute__((ext_vector_type(16))) _Float16 v16h;
typedef __attribute__((ext_vector_type(8)))  _Float16 v8h;
typedef __attribute__((ext_vector_type(8)))  float    v8f;

#define DIM   1024
#define SEQ   4096
#define BATCH 4
#define MROWS (BATCH*SEQ)   // 16384 token rows
#define CHUNK 128
#define NCHK  (SEQ/CHUNK)   // 32 chunks per sequence

// ---- CDNA5 async copy: global -> LDS, 16 bytes, tracked by ASYNCcnt ----
__device__ __forceinline__ void async_copy16(void* lds_dst, const void* gsrc) {
  uint32_t l = (uint32_t)(uintptr_t)lds_dst;   // flat LDS addr[31:0] == LDS byte offset
  asm volatile("global_load_async_to_lds_b128 %0, %1, off"
               :: "v"(l), "v"(gsrc) : "memory");
}
#define WAIT_ASYNCCNT(n) asm volatile("s_wait_asynccnt %0" :: "n"(n) : "memory")

// ---------------- fp32 -> fp16 weight conversion ----------------
__global__ void f32_to_f16_kernel(const float* __restrict__ src, _Float16* __restrict__ dst, int n) {
  int i = blockIdx.x * blockDim.x + threadIdx.x;
  if (i < n) dst[i] = (_Float16)src[i];
}

// ---------------- LayerNorm: one block (256 thr) per row of 1024 ----------------
__global__ void layernorm_f16_kernel(const float* __restrict__ X, const float* __restrict__ W,
                                     const float* __restrict__ B, _Float16* __restrict__ H) {
  __shared__ float s1[256];
  __shared__ float s2[256];
  const int row = blockIdx.x;
  const int tid = threadIdx.x;
  const float* xr = X + (size_t)row * DIM;
  float a0 = xr[tid], a1 = xr[tid + 256], a2 = xr[tid + 512], a3 = xr[tid + 768];
  s1[tid] = a0 + a1 + a2 + a3;
  s2[tid] = a0*a0 + a1*a1 + a2*a2 + a3*a3;
  __syncthreads();
  for (int off = 128; off > 0; off >>= 1) {
    if (tid < off) { s1[tid] += s1[tid + off]; s2[tid] += s2[tid + off]; }
    __syncthreads();
  }
  const float mean = s1[0] * (1.0f / DIM);
  const float var  = s2[0] * (1.0f / DIM) - mean * mean;
  const float rstd = rsqrtf(var + 1e-5f);
  _Float16* hr = H + (size_t)row * DIM;
  float av[4] = {a0, a1, a2, a3};
#pragma unroll
  for (int j = 0; j < 4; ++j) {
    int c = tid + j * 256;
    hr[c] = (_Float16)((av[j] - mean) * rstd * W[c] + B[c]);
  }
}

// ---------------- 3-pass chunked scan over the seq axis ----------------
// pass 1: per-chunk sums of g and g*v
__global__ void scan_partial_kernel(const float* __restrict__ G, const float* __restrict__ V,
                                    float* __restrict__ Sg, float* __restrict__ Sv) {
  const int gid = blockIdx.x * blockDim.x + threadIdx.x;   // 0 .. BATCH*NCHK*DIM-1
  const int f = gid & (DIM - 1);
  const int c = (gid >> 10) & (NCHK - 1);
  const int b = gid >> 15;
  const size_t base = ((size_t)b * SEQ + (size_t)c * CHUNK) * DIM + f;
  float sg = 0.0f, sv = 0.0f;
  for (int s = 0; s < CHUNK; ++s) {
    float gg = G[base + (size_t)s * DIM];
    float vv = V[base + (size_t)s * DIM];
    sg += gg;
    sv = fmaf(gg, vv, sv);
  }
  const size_t pidx = ((size_t)b * NCHK + c) * DIM + f;
  Sg[pidx] = sg; Sv[pidx] = sv;
}

// pass 2: exclusive prefix over chunks, per (batch, feature) column
__global__ void scan_prefix_kernel(float* __restrict__ Sg, float* __restrict__ Sv) {
  const int gid = blockIdx.x * blockDim.x + threadIdx.x;   // 0..4095
  const int b = gid >> 10, f = gid & (DIM - 1);
  float ag = 0.0f, av = 0.0f;
  for (int c = 0; c < NCHK; ++c) {
    size_t idx = ((size_t)b * NCHK + c) * DIM + f;
    float g0 = Sg[idx], v0 = Sv[idx];
    Sg[idx] = ag; Sv[idx] = av;
    ag += g0; av += v0;
  }
}

// pass 3: chunk-local scan with carry-in; x1 = x + mem
__global__ void scan_apply_kernel(const float* __restrict__ X, const float* __restrict__ G,
                                  const float* __restrict__ V, const float* __restrict__ Sg,
                                  const float* __restrict__ Sv, float* __restrict__ X1) {
  const int gid = blockIdx.x * blockDim.x + threadIdx.x;
  const int f = gid & (DIM - 1);
  const int c = (gid >> 10) & (NCHK - 1);
  const int b = gid >> 15;
  const size_t base = ((size_t)b * SEQ + (size_t)c * CHUNK) * DIM + f;
  const size_t pidx = ((size_t)b * NCHK + c) * DIM + f;
  float cg = Sg[pidx], cv = Sv[pidx];
  const float invS = 1.0f / (float)SEQ;
  for (int s = 0; s < CHUNK; ++s) {
    size_t idx = base + (size_t)s * DIM;
    float gg = G[idx];
    float vv = V[idx];
    cv = fmaf(gg, vv, cv);
    cg += gg;
    int pos = c * CHUNK + s + 1;
    float denom = sqrtf(cg + 1e-6f) * sqrtf((float)pos * invS + 0.1f);
    X1[idx] = X[idx] + cv / denom;
  }
}

// ---------------- WMMA GEMM:  out[m,n] = sum_k A[m,k] * W16[n,k]  (+bias, epilogue) ---------
// EPI: 0 = sigmoid -> f32,  1 = identity -> f32,  2 = exact GELU -> f16,  3 = +residual -> f32
template<int EPI>
__global__ void __launch_bounds__(256)
gemm_wmma_kernel(const _Float16* __restrict__ A, const _Float16* __restrict__ W16,
                 const float* __restrict__ bias, const float* __restrict__ res,
                 float* __restrict__ outF, _Float16* __restrict__ outH,
                 int M, int N, int K) {
  __shared__ __align__(16) _Float16 sA[2][128 * 32];
  __shared__ __align__(16) _Float16 sB[2][64 * 32];

  const int tid  = threadIdx.x;
  const int lane = tid & 31;
  const int wave = tid >> 5;        // 0..7 -> 16-row M strip
  const int hi   = lane >> 4;       // lane half selects K-phase (ISA frag layout)
  const int l15  = lane & 15;
  const int m0   = blockIdx.y * 128;
  const int n0   = blockIdx.x * 64;

  // Per-thread async staging assignment: 3 x 16B chunks per thread per tile
  // A tile = 128 rows x 32 halves = 512 chunks, B tile = 64 rows x 32 halves = 256 chunks.
  const int cA   = (tid & 3) * 8;
  const int rA0  = tid >> 2;
  const int rA1  = (tid + 256) >> 2;
  const int rB   = tid >> 2;
  const _Float16* gA0 = A   + (size_t)(m0 + rA0) * K + cA;
  const _Float16* gA1 = A   + (size_t)(m0 + rA1) * K + cA;
  const _Float16* gB  = W16 + (size_t)(n0 + rB)  * K + cA;

  v8f acc[4];
#pragma unroll
  for (int t = 0; t < 4; ++t)
#pragma unroll
    for (int r = 0; r < 8; ++r) acc[t][r] = 0.0f;

  // prologue: kick off tile 0 into buffer 0
  async_copy16(&sA[0][rA0 * 32 + cA], gA0);
  async_copy16(&sA[0][rA1 * 32 + cA], gA1);
  async_copy16(&sB[0][rB  * 32 + cA], gB);

  const int nsteps = K >> 5;
  for (int s = 0; s < nsteps; ++s) {
    const int buf = s & 1;
    if (s + 1 < nsteps) {
      // buffer buf^1 was last read at step s-1; the trailing barrier there makes it free
      const int k = (s + 1) << 5;
      async_copy16(&sA[buf ^ 1][rA0 * 32 + cA], gA0 + k);
      async_copy16(&sA[buf ^ 1][rA1 * 32 + cA], gA1 + k);
      async_copy16(&sB[buf ^ 1][rB  * 32 + cA], gB  + k);
      WAIT_ASYNCCNT(3);   // async loads complete in order => tile s (mine) has landed
    } else {
      WAIT_ASYNCCNT(0);
    }
    __syncthreads();      // all threads' tile-s DMA complete

    // A fragment (16x32 f16): lane<16 holds K {0..7,16..23}, lane>=16 holds K {8..15,24..31}
    const _Float16* ap = &sA[buf][(wave * 16 + l15) * 32];
    v8h alo = *(const v8h*)(ap + 8 * hi);
    v8h ahi = *(const v8h*)(ap + 16 + 8 * hi);
    v16h afrag = __builtin_shufflevector(alo, ahi, 0,1,2,3,4,5,6,7,8,9,10,11,12,13,14,15);

    v16h bfrag[4];
#pragma unroll
    for (int t = 0; t < 4; ++t) {
      const _Float16* bp = &sB[buf][(t * 16 + l15) * 32];
      v8h blo = *(const v8h*)(bp + 8 * hi);
      v8h bhi = *(const v8h*)(bp + 16 + 8 * hi);
      bfrag[t] = __builtin_shufflevector(blo, bhi, 0,1,2,3,4,5,6,7,8,9,10,11,12,13,14,15);
    }
#pragma unroll
    for (int t = 0; t < 4; ++t)
      acc[t] = __builtin_amdgcn_wmma_f32_16x16x32_f16(
          false, afrag, false, bfrag[t], (short)0, acc[t], false, false);

    __syncthreads();      // reads of buf done before step s+1 overwrites it
  }

  // epilogue + store (D layout: lanes 0-15 N=lane, M=r; lanes 16-31 N=lane-16, M=r+8)
#pragma unroll
  for (int t = 0; t < 4; ++t) {
    int ncol = n0 + t * 16 + l15;
    float bv = bias[ncol];
#pragma unroll
    for (int r = 0; r < 8; ++r) {
      int mrow = m0 + wave * 16 + hi * 8 + r;
      size_t idx = (size_t)mrow * N + ncol;
      float val = acc[t][r] + bv;
      if (EPI == 0) {
        outF[idx] = 1.0f / (1.0f + expf(-val));
      } else if (EPI == 1) {
        outF[idx] = val;
      } else if (EPI == 2) {
        outH[idx] = (_Float16)(0.5f * val * (1.0f + erff(val * 0.70710678118654752f)));
      } else {
        outF[idx] = res[idx] + val;
      }
    }
  }
}

extern "C" void kernel_launch(void* const* d_in, const int* in_sizes, int n_in,
                              void* d_out, int out_size, void* d_ws, size_t ws_size,
                              hipStream_t stream) {
  (void)in_sizes; (void)n_in; (void)out_size; (void)ws_size;
  const float* x      = (const float*)d_in[0];
  const float* ln1_w  = (const float*)d_in[1];
  const float* ln1_b  = (const float*)d_in[2];
  const float* ln2_w  = (const float*)d_in[3];
  const float* ln2_b  = (const float*)d_in[4];
  const float* gate_w = (const float*)d_in[5];
  const float* gate_b = (const float*)d_in[6];
  const float* val_w  = (const float*)d_in[7];
  const float* val_b  = (const float*)d_in[8];
  const float* ffn1_w = (const float*)d_in[9];
  const float* ffn1_b = (const float*)d_in[10];
  const float* ffn2_w = (const float*)d_in[11];
  const float* ffn2_b = (const float*)d_in[12];

  char* ws = (char*)d_ws;
  const size_t MB = 1024ull * 1024ull;
  _Float16* h16   = (_Float16*)(ws);             // 32 MB: LN output (reused for LN2)
  float*    G     = (float*)(ws + 32 * MB);      // 64 MB
  float*    V     = (float*)(ws + 96 * MB);      // 64 MB
  _Float16* FF1   = (_Float16*)(ws + 32 * MB);   // 128 MB, reuses dead G/V region
  _Float16* gw16  = (_Float16*)(ws + 160 * MB);  // 2 MB
  _Float16* vw16  = (_Float16*)(ws + 162 * MB);  // 2 MB
  _Float16* w1_16 = (_Float16*)(ws + 164 * MB);  // 8 MB
  _Float16* w2_16 = (_Float16*)(ws + 172 * MB);  // 8 MB
  float*    Sg    = (float*)(ws + 180 * MB);     // 512 KB chunk partials
  float*    Sv    = (float*)(ws + 181 * MB);     // 512 KB chunk partials
  float*    X1    = (float*)d_out;               // x + mem lives in d_out

  // 1) weights -> f16
  f32_to_f16_kernel<<<(DIM * DIM) / 256, 256, 0, stream>>>(gate_w, gw16, DIM * DIM);
  f32_to_f16_kernel<<<(DIM * DIM) / 256, 256, 0, stream>>>(val_w,  vw16, DIM * DIM);
  f32_to_f16_kernel<<<(4 * DIM * DIM) / 256, 256, 0, stream>>>(ffn1_w, w1_16, 4 * DIM * DIM);
  f32_to_f16_kernel<<<(4 * DIM * DIM) / 256, 256, 0, stream>>>(ffn2_w, w2_16, 4 * DIM * DIM);

  // 2) h = LN1(x)
  layernorm_f16_kernel<<<MROWS, 256, 0, stream>>>(x, ln1_w, ln1_b, h16);

  // 3) g = sigmoid(h @ gate_w^T + b), v = h @ val_w^T + b
  gemm_wmma_kernel<0><<<dim3(DIM / 64, MROWS / 128), 256, 0, stream>>>(
      h16, gw16, gate_b, nullptr, G, nullptr, MROWS, DIM, DIM);
  gemm_wmma_kernel<1><<<dim3(DIM / 64, MROWS / 128), 256, 0, stream>>>(
      h16, vw16, val_b, nullptr, V, nullptr, MROWS, DIM, DIM);

  // 4) x1 = x + cumsum(g*v) / (sqrt(cumsum(g)+1e-6)*sqrt(pos/S+0.1))   [3-pass scan]
  scan_partial_kernel<<<(BATCH * NCHK * DIM) / 256, 256, 0, stream>>>(G, V, Sg, Sv);
  scan_prefix_kernel<<<(BATCH * DIM) / 256, 256, 0, stream>>>(Sg, Sv);
  scan_apply_kernel<<<(BATCH * NCHK * DIM) / 256, 256, 0, stream>>>(x, G, V, Sg, Sv, X1);

  // 5) h2 = LN2(x1)
  layernorm_f16_kernel<<<MROWS, 256, 0, stream>>>(X1, ln2_w, ln2_b, h16);

  // 6) ff1 = gelu(h2 @ ffn1_w^T + b)  -> f16
  gemm_wmma_kernel<2><<<dim3(4 * DIM / 64, MROWS / 128), 256, 0, stream>>>(
      h16, w1_16, ffn1_b, nullptr, nullptr, FF1, MROWS, 4 * DIM, DIM);

  // 7) out = x1 + (ff1 @ ffn2_w^T + b)
  gemm_wmma_kernel<3><<<dim3(DIM / 64, MROWS / 128), 256, 0, stream>>>(
      FF1, w2_16, ffn2_b, X1, X1, nullptr, MROWS, DIM, 4 * DIM);
}
